// gnn_network_23459111370852
// MI455X (gfx1250) — matrix operand order repaired and checked by reading, not compiled, not run on hardware
//
#include <hip/hip_runtime.h>
#include <hip/hip_bf16.h>
#include <stdint.h>

// Problem dimensions (fixed by the reference).
#define N_NODES 65536
#define E_EDGES 1048576
#define D_CAT   1024   // 256 + 768 fused input features
#define D_IN    512
#define D_H     512
#define D_OUT   128
#define S_SAMP  256

typedef __attribute__((ext_vector_type(16))) __bf16 bf16x16;
typedef __attribute__((ext_vector_type(8)))  float  floatx8;

__device__ __forceinline__ unsigned short f2bf(float f) {
  unsigned int u = __float_as_uint(f);
  return (unsigned short)((u + 0x7FFFu + ((u >> 16) & 1u)) >> 16);  // RNE
}

__device__ __forceinline__ int imin(int a, int b) { return a < b ? a : b; }

// ---------------------------------------------------------------------------
// WMMA fragment layouts used here (contiguous k, not the ISA table order; wave32, 16-bit data):
// A (16x32): lane l -> row m = l&15; element i -> k = 16*(l>>4)+i
// B (32x16): lane l -> col n = l&15; element i -> k = 16*(l>>4)+i
// Packed storage: [tileRow][tileK][lane][16] bf16, 32 B per lane per fragment.
// ---------------------------------------------------------------------------

// Map (m, k) of a row-major matrix with row-tile mt and ktp k-tiles to the
// flat packed-A element index.
__device__ __forceinline__ size_t a_elem_index(int mt, int ktp, int m, int k) {
  int lane = ((((k >> 4) & 1) << 4) | (m & 15));
  int i    = (k & 15);
  return ((size_t)(mt * ktp + (k >> 5)) * 32 + lane) * 16 + i;
}

__global__ __launch_bounds__(256)
void pack_a(const float* __restrict__ src, unsigned short* __restrict__ dst,
            int M, int K) {
  int o = blockIdx.x * blockDim.x + threadIdx.x;
  if (o >= M * K) return;
  int i = o & 15, lane = (o >> 4) & 31, tile = o >> 9;
  int ktiles = K >> 5;
  int kt = tile % ktiles, mt = tile / ktiles;
  int m = (mt << 4) + (lane & 15);
  int k = (kt << 5) + ((lane >> 4) << 4) + i;
  dst[o] = f2bf(src[(size_t)m * K + k]);
}

// Pack row-major W[K,N] into B-fragment layout [kt][nt][lane][16].
__global__ __launch_bounds__(256)
void pack_b(const float* __restrict__ src, unsigned short* __restrict__ dst,
            int K, int N) {
  int o = blockIdx.x * blockDim.x + threadIdx.x;
  if (o >= K * N) return;
  int i = o & 15, lane = (o >> 4) & 31, tile = o >> 9;
  int ntiles = N >> 4;
  int nt = tile % ntiles, kt = tile / ntiles;
  int k = (kt << 5) + ((lane >> 4) << 4) + i;
  int n = (nt << 4) + (lane & 15);
  dst[o] = f2bf(src[(size_t)k * N + n]);
}

// Pack concat(Wa[Ka,N]; Wb[K-Ka,N]) into B-fragment layout.
__global__ __launch_bounds__(256)
void pack_b_concat(const float* __restrict__ Wa, const float* __restrict__ Wb,
                   int Ka, unsigned short* __restrict__ dst, int K, int N) {
  int o = blockIdx.x * blockDim.x + threadIdx.x;
  if (o >= K * N) return;
  int i = o & 15, lane = (o >> 4) & 31, tile = o >> 9;
  int ntiles = N >> 4;
  int nt = tile % ntiles, kt = tile / ntiles;
  int k = (kt << 5) + ((lane >> 4) << 4) + i;
  int n = (nt << 4) + (lane & 15);
  float v = (k < Ka) ? Wa[(size_t)k * N + n] : Wb[(size_t)(k - Ka) * N + n];
  dst[o] = f2bf(v);
}

// ---------------------------------------------------------------------------
// Fused bias(+ReLU) epilogues that emit packed fragments directly.
// ---------------------------------------------------------------------------
__global__ __launch_bounds__(256)
void bias_act_pack_a(const float* __restrict__ src, const float* __restrict__ bias,
                     unsigned short* __restrict__ dst, int M, int K, int relu) {
  int o = blockIdx.x * blockDim.x + threadIdx.x;
  if (o >= M * K) return;
  int i = o & 15, lane = (o >> 4) & 31, tile = o >> 9;
  int ktiles = K >> 5;
  int kt = tile % ktiles, mt = tile / ktiles;
  int m = (mt << 4) + (lane & 15);
  int k = (kt << 5) + ((lane >> 4) << 4) + i;
  float v = src[(size_t)m * K + k] + bias[k];
  if (relu) v = fmaxf(v, 0.0f);
  dst[o] = f2bf(v);
}

// B^T operand: B[k][n] = src[n][k] + bias[k]; layout [rt][kt][lane][16].
__global__ __launch_bounds__(256)
void bias_pack_bt(const float* __restrict__ src, const float* __restrict__ bias,
                  unsigned short* __restrict__ dst, int R, int K) {
  int o = blockIdx.x * blockDim.x + threadIdx.x;
  if (o >= R * K) return;
  int i = o & 15, lane = (o >> 4) & 31, tile = o >> 9;
  int ktiles = K >> 5;
  int kt = tile % ktiles, rt = tile / ktiles;
  int r = (rt << 4) + (lane & 15);
  int k = (kt << 5) + ((lane >> 4) << 4) + i;
  dst[o] = f2bf(src[(size_t)r * K + k] + bias[k]);
}

// ---------------------------------------------------------------------------
// Double-buffered WMMA GEMM core: 16x64 C tile per wave, K-loop unrolled x2.
// Even iterations consume buffer 0, odd consume buffer 1; each buffer is
// refilled immediately after its WMMA group, so load latency overlaps the
// other buffer's WMMAs without any register-copy rotation.
// Requires ktiles even (true for all stages here: 32, 16, 4).
// ---------------------------------------------------------------------------
#define WMMA_BF16(a, b, c) \
  __builtin_amdgcn_wmma_f32_16x16x32_bf16(false, (a), false, (b), (short)0, (c), false, false)

__device__ __forceinline__ void gemm4_core(const bf16x16* __restrict__ A,
                                           const bf16x16* __restrict__ B,
                                           int ktiles, int ntiles, int nt0, int lane,
                                           floatx8& c0, floatx8& c1,
                                           floatx8& c2, floatx8& c3) {
  const bf16x16* B0 = B + (size_t)nt0 * 32;
  const bf16x16* B1 = B + ((size_t)ntiles + nt0) * 32;
  bf16x16 a0  = A[lane];
  bf16x16 b00 = B0[lane], b01 = B0[32 + lane], b02 = B0[64 + lane], b03 = B0[96 + lane];
  bf16x16 a1  = A[32 + lane];
  bf16x16 b10 = B1[lane], b11 = B1[32 + lane], b12 = B1[64 + lane], b13 = B1[96 + lane];
  for (int kt = 0; kt < ktiles; kt += 2) {
    // Far-ahead prefetch of the A stream (clamped; branch-free).
    __builtin_prefetch(A + (size_t)imin(kt + 8, ktiles - 1) * 32 + lane, 0, 3);
    c0 = WMMA_BF16(a0, b00, c0);
    c1 = WMMA_BF16(a0, b01, c1);
    c2 = WMMA_BF16(a0, b02, c2);
    c3 = WMMA_BF16(a0, b03, c3);
    {   // refill buffer 0 with next even k-tile (clamped at tail, never used)
      int ke = imin(kt + 2, ktiles - 2);
      const bf16x16* Be = B + ((size_t)ke * ntiles + nt0) * 32;
      a0  = A[(size_t)ke * 32 + lane];
      b00 = Be[lane]; b01 = Be[32 + lane]; b02 = Be[64 + lane]; b03 = Be[96 + lane];
    }
    c0 = WMMA_BF16(a1, b10, c0);
    c1 = WMMA_BF16(a1, b11, c1);
    c2 = WMMA_BF16(a1, b12, c2);
    c3 = WMMA_BF16(a1, b13, c3);
    {   // refill buffer 1 with next odd k-tile (clamped at tail)
      int ko = imin(kt + 3, ktiles - 1);
      const bf16x16* Bo = B + ((size_t)ko * ntiles + nt0) * 32;
      a1  = A[(size_t)ko * 32 + lane];
      b10 = Bo[lane]; b11 = Bo[32 + lane]; b12 = Bo[64 + lane]; b13 = Bo[96 + lane];
    }
  }
}

// fp32 row-major output + optional dual bias.
__global__ __launch_bounds__(32)
void wmma_gemm4(const unsigned short* __restrict__ Apack,
                const unsigned short* __restrict__ Bpack,
                float* __restrict__ C, int M, int K, int Nn,
                const float* __restrict__ bias0, const float* __restrict__ bias1) {
  int ngroups = Nn >> 6;
  int tile = blockIdx.x;
  int mt  = tile / ngroups;
  int nt0 = (tile % ngroups) << 2;
  int lane = threadIdx.x;
  int ktiles = K >> 5, ntiles = Nn >> 4;
  const bf16x16* A = reinterpret_cast<const bf16x16*>(Apack) + (size_t)mt * ktiles * 32;
  const bf16x16* B = reinterpret_cast<const bf16x16*>(Bpack);
  floatx8 c0 = {}, c1 = {}, c2 = {}, c3 = {};
  gemm4_core(A, B, ktiles, ntiles, nt0, lane, c0, c1, c2, c3);
  int mbase = (mt << 4) + ((lane >> 4) << 3);
  floatx8 acc[4] = {c0, c1, c2, c3};
#pragma unroll
  for (int q = 0; q < 4; ++q) {
    int n = ((nt0 + q) << 4) + (lane & 15);
    float badd = 0.0f;
    if (bias0) badd += bias0[n];
    if (bias1) badd += bias1[n];
#pragma unroll
    for (int r = 0; r < 8; ++r)
      C[(size_t)(mbase + r) * Nn + n] = acc[q][r] + badd;
  }
}

// Same GEMM, but stores the result directly as packed-A bf16 fragments for the
// next GEMM (K' = Nn, ktp = Nn/32). Skips the fp32 intermediate entirely.
__global__ __launch_bounds__(32)
void wmma_gemm4_packout(const unsigned short* __restrict__ Apack,
                        const unsigned short* __restrict__ Bpack,
                        unsigned short* __restrict__ D, int M, int K, int Nn,
                        const float* __restrict__ bias0, const float* __restrict__ bias1) {
  int ngroups = Nn >> 6;
  int tile = blockIdx.x;
  int mt  = tile / ngroups;
  int nt0 = (tile % ngroups) << 2;
  int lane = threadIdx.x;
  int ktiles = K >> 5, ntiles = Nn >> 4, ktp = Nn >> 5;
  const bf16x16* A = reinterpret_cast<const bf16x16*>(Apack) + (size_t)mt * ktiles * 32;
  const bf16x16* B = reinterpret_cast<const bf16x16*>(Bpack);
  floatx8 c0 = {}, c1 = {}, c2 = {}, c3 = {};
  gemm4_core(A, B, ktiles, ntiles, nt0, lane, c0, c1, c2, c3);
  floatx8 acc[4] = {c0, c1, c2, c3};
#pragma unroll
  for (int q = 0; q < 4; ++q) {
    int n = ((nt0 + q) << 4) + (lane & 15);
    float badd = 0.0f;
    if (bias0) badd += bias0[n];
    if (bias1) badd += bias1[n];
#pragma unroll
    for (int r = 0; r < 8; ++r) {
      int m = (mt << 4) + r + ((lane >> 4) << 3);
      D[a_elem_index(mt, ktp, m, n)] = f2bf(acc[q][r] + badd);
    }
  }
}

// ---------------------------------------------------------------------------
// Bilinear scorer: per batch b, scores_b = t_b(256x128) @ h_b^T(128x256),
// 4 j-tiles per wave, fully unrolled K-loop (ktiles = 4, compile-time),
// fused Linear(1,2) epilogue with float2 stores.
// ---------------------------------------------------------------------------
__global__ __launch_bounds__(32)
void wmma_bilinear4(const unsigned short* __restrict__ TpackA,
                    const unsigned short* __restrict__ HpackBT,
                    float* __restrict__ out,
                    const float* __restrict__ lin_w, const float* __restrict__ lin_b) {
  int tile = blockIdx.x;                 // b*64 + mt*4 + jg
  int jg = tile & 3;
  int mt = (tile >> 2) & 15;
  int b  = tile >> 6;
  int lane = threadIdx.x;
  const int ktiles = D_OUT >> 5;         // 4
  int mtg = b * (S_SAMP >> 4) + mt;
  int nt0 = b * (S_SAMP >> 4) + (jg << 2);
  const bf16x16* A = reinterpret_cast<const bf16x16*>(TpackA) + (size_t)mtg * ktiles * 32;
  const bf16x16* B = reinterpret_cast<const bf16x16*>(HpackBT);
  floatx8 c0 = {}, c1 = {}, c2 = {}, c3 = {};
#pragma unroll
  for (int kt = 0; kt < ktiles; ++kt) {
    bf16x16 a  = A[kt * 32 + lane];
    bf16x16 b0 = B[((size_t)(nt0 + 0) * ktiles + kt) * 32 + lane];
    bf16x16 b1 = B[((size_t)(nt0 + 1) * ktiles + kt) * 32 + lane];
    bf16x16 b2 = B[((size_t)(nt0 + 2) * ktiles + kt) * 32 + lane];
    bf16x16 b3 = B[((size_t)(nt0 + 3) * ktiles + kt) * 32 + lane];
    c0 = WMMA_BF16(a, b0, c0);
    c1 = WMMA_BF16(a, b1, c1);
    c2 = WMMA_BF16(a, b2, c2);
    c3 = WMMA_BF16(a, b3, c3);
  }
  float w0 = lin_w[0], w1 = lin_w[1], d0 = lin_b[0], d1 = lin_b[1];
  int ibase = (mt << 4) + ((lane >> 4) << 3);
  floatx8 acc[4] = {c0, c1, c2, c3};
#pragma unroll
  for (int q = 0; q < 4; ++q) {
    int j = ((jg << 2) + q) * 16 + (lane & 15);
#pragma unroll
    for (int r = 0; r < 8; ++r) {
      float s = acc[q][r];
      size_t idx = ((((size_t)b * S_SAMP + ibase + r) * S_SAMP) + j) * 2;
      float2 v; v.x = s * w0 + d0; v.y = s * w1 + d1;
      *(float2*)(out + idx) = v;
    }
  }
}

// ---------------------------------------------------------------------------
// GCN aggregation: symmetric deg^-1/2 normalization + scatter-add.
// ---------------------------------------------------------------------------
__global__ __launch_bounds__(256)
void deg_init(float* __restrict__ deg) {
  int i = blockIdx.x * blockDim.x + threadIdx.x;
  if (i < N_NODES) deg[i] = 1.0f;  // self loop
}

__global__ __launch_bounds__(256)
void deg_count(const long long* __restrict__ dstv, float* __restrict__ deg) {
  int e = blockIdx.x * blockDim.x + threadIdx.x;
  if (e < E_EDGES) atomicAdd(&deg[(int)dstv[e]], 1.0f);
}

__global__ __launch_bounds__(256)
void deg_rsqrt(float* __restrict__ deg) {
  int i = blockIdx.x * blockDim.x + threadIdx.x;
  if (i < N_NODES) deg[i] = rsqrtf(deg[i]);
}

// acc[i,f] = tmp[i,f] * inv[i]^2   (self-loop term; initializes acc)
__global__ __launch_bounds__(256)
void agg_self(const float* __restrict__ tmp, const float* __restrict__ inv,
              float* __restrict__ acc, int F) {
  size_t idx = (size_t)blockIdx.x * blockDim.x + threadIdx.x;
  if (idx >= (size_t)N_NODES * F) return;
  int node = (int)(idx / F);
  float w = inv[node];
  acc[idx] = tmp[idx] * w * w;
}

// acc[d,:] += tmp[s,:] * inv[s]*inv[d]; one block per edge.
__global__ void agg_edges(const float* __restrict__ tmp,
                          const long long* __restrict__ srcv,
                          const long long* __restrict__ dstv,
                          const float* __restrict__ inv,
                          float* __restrict__ acc, int F) {
  int e = blockIdx.x;
  int s = (int)srcv[e];
  int d = (int)dstv[e];
  float w = inv[s] * inv[d];
  const float* ps = tmp + (size_t)s * F;
  float*       pd = acc + (size_t)d * F;
  for (int f = threadIdx.x; f < F; f += blockDim.x)
    atomicAdd(&pd[f], ps[f] * w);
}

// ---------------------------------------------------------------------------
extern "C" void kernel_launch(void* const* d_in, const int* in_sizes, int n_in,
                              void* d_out, int out_size, void* d_ws, size_t ws_size,
                              hipStream_t stream) {
  const float* x   = (const float*)d_in[0];
  const float* Wt1 = (const float*)d_in[1];
  const float* bt1 = (const float*)d_in[2];
  const float* Wt2 = (const float*)d_in[3];
  const float* bt2 = (const float*)d_in[4];
  const float* W1  = (const float*)d_in[5];
  const float* b1  = (const float*)d_in[6];
  const float* W2  = (const float*)d_in[7];
  const float* b2  = (const float*)d_in[8];
  const float* Mx  = (const float*)d_in[9];
  const float* lw  = (const float*)d_in[10];
  const float* lb  = (const float*)d_in[11];
  const long long* ei   = (const long long*)d_in[12];
  const long long* esrc = ei;
  const long long* edst = ei + E_EDGES;
  float* out = (float*)d_out;

  const size_t MB = 1024ull * 1024ull;
  if (ws_size < 260 * MB) return;  // scratch plan needs 260 MB

  uint8_t* R1 = (uint8_t*)d_ws;    // 128 MB, aliased across stages
  uint8_t* R2 = R1 + 128 * MB;     // 128 MB, aliased across stages
  uint8_t* R3 = R1 + 256 * MB;     // small persistent region
  float*          inv = (float*)R3;                                  // 256 KB
  unsigned short* W0p = (unsigned short*)(R3 + 1 * MB);              // 1 MB
  unsigned short* W1p = (unsigned short*)(R3 + 2 * MB);              // 512 KB
  unsigned short* W2p = (unsigned short*)(R3 + 3 * MB);              // 128 KB
  unsigned short* Mp  = (unsigned short*)(R3 + 3 * MB + 512 * 1024); // 32 KB

  // --- weight packing (B-fragment layouts) ---
  pack_b_concat<<<(D_CAT * D_IN) / 256, 256, 0, stream>>>(Wt1, Wt2, 256, W0p, D_CAT, D_IN);
  pack_b<<<(D_IN * D_H)    / 256, 256, 0, stream>>>(W1, W1p, D_IN, D_H);
  pack_b<<<(D_H * D_OUT)   / 256, 256, 0, stream>>>(W2, W2p, D_H, D_OUT);
  pack_b<<<(D_OUT * D_OUT) / 256, 256, 0, stream>>>(Mx, Mp, D_OUT, D_OUT);

  // --- degrees: deg = 1 + scatter(ones, dst); inv = rsqrt(deg) ---
  deg_init<<<N_NODES / 256, 256, 0, stream>>>(inv);
  deg_count<<<E_EDGES / 256, 256, 0, stream>>>(edst, inv);
  deg_rsqrt<<<N_NODES / 256, 256, 0, stream>>>(inv);

  // --- stage 1: h0 = x @ Wcat + b_t1 + b_t2, stored directly as packed A ---
  unsigned short* PX  = (unsigned short*)R1;  // 128 MB packed x
  unsigned short* PA1 = (unsigned short*)R2;  // 64 MB packed h0
  pack_a<<<(N_NODES * D_CAT) / 256, 256, 0, stream>>>(x, PX, N_NODES, D_CAT);
  wmma_gemm4_packout<<<(N_NODES / 16) * (D_IN / 64), 32, 0, stream>>>(
      PX, W0p, PA1, N_NODES, D_CAT, D_IN, bt1, bt2);

  // --- stage 2: GCN1  t1 = h0 @ W1 -> scatter -> +b1, ReLU -> packed A ---
  float* T1 = (float*)R1;                     // 128 MB (PX dead)
  wmma_gemm4<<<(N_NODES / 16) * (D_H / 64), 32, 0, stream>>>(
      PA1, W1p, T1, N_NODES, D_IN, D_H, nullptr, nullptr);
  float* ACC1 = (float*)R2;                   // 128 MB (PA1 dead)
  agg_self<<<(N_NODES * D_H) / 256, 256, 0, stream>>>(T1, inv, ACC1, D_H);
  agg_edges<<<E_EDGES, 128, 0, stream>>>(T1, esrc, edst, inv, ACC1, D_H);
  unsigned short* PA2 = (unsigned short*)R1;  // 64 MB (T1 dead)
  bias_act_pack_a<<<(N_NODES * D_H) / 256, 256, 0, stream>>>(
      ACC1, b1, PA2, N_NODES, D_H, 1);

  // --- stage 3: GCN2  t2 = h1 @ W2 -> scatter -> +b2, packed A and B^T ---
  float* T2 = (float*)(R1 + 64 * MB);         // 32 MB
  wmma_gemm4<<<(N_NODES / 16) * (D_OUT / 64), 32, 0, stream>>>(
      PA2, W2p, T2, N_NODES, D_H, D_OUT, nullptr, nullptr);
  float* ACC2 = (float*)(R1 + 96 * MB);       // 32 MB
  agg_self<<<(N_NODES * D_OUT) / 256, 256, 0, stream>>>(T2, inv, ACC2, D_OUT);
  agg_edges<<<E_EDGES, 64, 0, stream>>>(T2, esrc, edst, inv, ACC2, D_OUT);

  unsigned short* PHA = (unsigned short*)R2;             // 32 MB (ACC1 dead)
  unsigned short* PHB = (unsigned short*)(R2 + 32 * MB); // 32 MB
  bias_act_pack_a<<<(N_NODES * D_OUT) / 256, 256, 0, stream>>>(
      ACC2, b2, PHA, N_NODES, D_OUT, 0);
  bias_pack_bt<<<(N_NODES * D_OUT) / 256, 256, 0, stream>>>(
      ACC2, b2, PHB, N_NODES, D_OUT);

  // --- stage 4: t = h @ matrix (packed-A out), then scores + Linear(1,2) ---
  unsigned short* PTA = (unsigned short*)(R2 + 64 * MB); // 32 MB
  wmma_gemm4_packout<<<(N_NODES / 16) * (D_OUT / 64), 32, 0, stream>>>(
      PHA, Mp, PTA, N_NODES, D_OUT, D_OUT, nullptr, nullptr);

  wmma_bilinear4<<<(N_NODES / S_SAMP) * (S_SAMP / 16) * (S_SAMP / 64), 32, 0, stream>>>(
      PTA, PHB, out, lw, lb);
}
